// PointerDecoder_63307817943535
// MI455X (gfx1250) — matrix operand order repaired and checked by reading, not compiled, not run on hardware
//
#include <hip/hip_runtime.h>
#include <hip/hip_bf16.h>

// ---------------------------------------------------------------------------
// PointerDecoder for MI455X (gfx1250, wave32).
//  - e_g / e_p precomputed once via bf16 WMMA, stored bf16 (67MB -> L2 resident)
//  - per-step GEMMs (LSTM gates, qq projections) via v_wmma_f32_16x16x32_bf16
//  - attention / softmax / LSTM pointwise in f32
//  - LDS A-tile staging: one float2 -> packed bf16 pair per thread (vector path)
// Workspace requirement: ~73 MB.
// ---------------------------------------------------------------------------

#define BB 1024   // batch
#define LL 128    // seq len == decode steps
#define EE 128    // embed
#define HH 128    // hidden

typedef __attribute__((ext_vector_type(16))) __bf16 v16bf;
typedef __attribute__((ext_vector_type(2)))  __bf16 v2bf;
typedef __attribute__((ext_vector_type(8)))  float  v8f;

__device__ __forceinline__ v8f wmma_bf16(v16bf a, v16bf b, v8f c) {
  // (neg_a, A, neg_b, B, c_mod, C, reuse_a, reuse_b)
  return __builtin_amdgcn_wmma_f32_16x16x32_bf16(false, a, false, b, (short)0, c,
                                                 false, false);
}

// A-matrix 16x32 bf16 fragment from an LDS tile laid out row-major [16][32].
// Lanes 0-15 row M=lane hold K=0..15; lanes 16-31 hold K=16..31 of row M=lane-16
// (contiguous, matching load_frag_B).
__device__ __forceinline__ v16bf load_frag_A(const __bf16* tile, int lane) {
  int m = lane & 15, kh = (lane >> 4) & 1;
  const __bf16* row = tile + m * 32;
  v16bf a;
#pragma unroll
  for (int j = 0; j < 8; ++j) {
    int k = 16 * kh + 2 * j;
    a[2 * j]     = row[k];
    a[2 * j + 1] = row[k + 1];
  }
  return a;
}

// B-matrix 32x16 bf16 fragment with B[k][n] = W[n0+n][k0+k]; lane holds column
// n = lane&15, lanes 0-15 cover K=0..15, lanes 16-31 K=16..31 (contiguous 32B).
__device__ __forceinline__ v16bf load_frag_B(const __bf16* W, int ldk, int n0,
                                             int k0, int lane) {
  int n = lane & 15, kh = (lane >> 4) & 1;
  const __bf16* row = W + (size_t)(n0 + n) * ldk + k0 + kh * 16;
  v16bf b;
#pragma unroll
  for (int j = 0; j < 16; ++j) b[j] = row[j];
  return b;
}

// Stage one float2 (8B-aligned) into LDS as a packed bf16 pair (4B store).
__device__ __forceinline__ void stage_pair(__bf16* dst, const float* src) {
  float2 v2 = *(const float2*)src;
  v2bf p;
  p[0] = (__bf16)v2.x;
  p[1] = (__bf16)v2.y;
  *(v2bf*)dst = p;
}

__device__ __forceinline__ float sigf(float x) { return 1.0f / (1.0f + __expf(-x)); }

// ---------------------------------------------------------------------------
// Setup: repack weights to bf16, init h/c state, zero mask.
// ---------------------------------------------------------------------------
__global__ void setup_kernel(const float* W_ih, const float* W_hh, const float* gWq,
                             const float* pWq, const float* gWref, const float* pWref,
                             const float* h0, const float* c0,
                             __bf16* Wcat, __bf16* gWqb, __bf16* pWqb,
                             __bf16* gWrefb, __bf16* pWrefb,
                             float* h, float* c, unsigned char* mask) {
  int g = blockIdx.x * blockDim.x + threadIdx.x;
  if (g < 512 * 256) {                     // Wcat[n][k] = [W_ih | W_hh]
    int n = g >> 8, k = g & 255;
    Wcat[g] = (__bf16)((k < 128) ? W_ih[n * 128 + k] : W_hh[n * 128 + (k - 128)]);
    return;
  }
  g -= 512 * 256;
  if (g < 16384) { gWqb[g] = (__bf16)gWq[g]; return; }   g -= 16384;
  if (g < 16384) { pWqb[g] = (__bf16)pWq[g]; return; }   g -= 16384;
  if (g < 16384) { gWrefb[g] = (__bf16)gWref[g]; return; } g -= 16384;
  if (g < 16384) { pWrefb[g] = (__bf16)pWref[g]; return; } g -= 16384;
  if (g < BB * HH) { h[g] = h0[g]; return; }  g -= BB * HH;
  if (g < BB * HH) { c[g] = c0[g]; return; }  g -= BB * HH;
  if (g < BB * LL) { mask[g] = 0; return; }
}

// ---------------------------------------------------------------------------
// Hoisted reference projections: e[b][l][o] = sum_h Wref[o][h]*ctx[l][b][h] + bias[o]
// grid (B/16, L, 2); block 256 (8 waves, each owns a 16-col N-tile of N=128).
// ---------------------------------------------------------------------------
__global__ void ref_proj_kernel(const float* context, const __bf16* Wg, const float* bg,
                                const __bf16* Wp, const float* bp,
                                __bf16* eg, __bf16* ep) {
  __shared__ __bf16 At[16 * 32];
  int b0 = blockIdx.x * 16, l = blockIdx.y;
  const __bf16* W   = blockIdx.z ? Wp : Wg;
  const float* bias = blockIdx.z ? bp : bg;
  __bf16* e         = blockIdx.z ? ep : eg;
  int t = threadIdx.x, lane = t & 31, wave = t >> 5;
  int n0 = wave * 16;
  int sm = t >> 4, scol = (t & 15) * 2;    // staging: row / col-pair
  v8f acc = {};
  for (int k0 = 0; k0 < HH; k0 += 32) {
    __syncthreads();
    stage_pair(At + sm * 32 + scol,
               &context[((size_t)l * BB + (b0 + sm)) * HH + k0 + scol]);
    __syncthreads();
    v16bf a = load_frag_A(At, lane);
    v16bf b = load_frag_B(W, HH, n0, k0, lane);
    acc = wmma_bf16(a, b, acc);
  }
  int n = n0 + (lane & 15), kh = lane >> 4;
  float bs = bias[n];
#pragma unroll
  for (int r = 0; r < 8; ++r) {
    int m = r + 8 * kh;
    e[((size_t)(b0 + m) * LL + l) * HH + n] = (__bf16)(acc[r] + bs);
  }
}

// ---------------------------------------------------------------------------
// LSTM gates GEMM: gates[b][n] = [x|h][b][:] @ Wcat[n][:] + b_ih[n] + b_hh[n]
// x row gathered per step. grid (B/16, 4); block 256.
// ---------------------------------------------------------------------------
__global__ void lstm_gates_kernel(const float* dec_in, const float* emb,
                                  const int* tours, const float* h,
                                  const __bf16* Wcat, const float* b_ih,
                                  const float* b_hh, float* gates, int s) {
  __shared__ __bf16 At[16 * 32];
  __shared__ int rowidx[16];
  int b0 = blockIdx.x * 16;
  int t = threadIdx.x, lane = t & 31, wave = t >> 5;
  int n0 = blockIdx.y * 128 + wave * 16;
  if (t < 16) rowidx[t] = (s == 0) ? -1 : tours[(b0 + t) * LL + (s - 1)];
  int sm = t >> 4, scol = (t & 15) * 2, b = b0 + sm;
  v8f acc = {};
  for (int k0 = 0; k0 < 256; k0 += 32) {
    __syncthreads();
    {
      int kk = k0 + scol;                  // pair never straddles the x|h split
      const float* src;
      if (kk < 128) {
        int ri = rowidx[sm];
        src = (ri < 0) ? &dec_in[b * EE + kk]
                       : &emb[((size_t)ri * BB + b) * EE + kk];
      } else {
        src = &h[b * HH + (kk - 128)];
      }
      stage_pair(At + sm * 32 + scol, src);
    }
    __syncthreads();
    v16bf a = load_frag_A(At, lane);
    v16bf bb = load_frag_B(Wcat, 256, n0, k0, lane);
    acc = wmma_bf16(a, bb, acc);
  }
  int n = n0 + (lane & 15), kh = lane >> 4;
  float bs = b_ih[n] + b_hh[n];
#pragma unroll
  for (int r = 0; r < 8; ++r)
    gates[(size_t)(b0 + r + 8 * kh) * 512 + n] = acc[r] + bs;
}

// LSTM pointwise: gate order i,f,g,o; updates h,c in place.
__global__ void lstm_pointwise_kernel(const float* gates, float* h, float* c) {
  int g = blockIdx.x * blockDim.x + threadIdx.x;   // B*H
  int b = g >> 7, n = g & 127;
  const float* gr = gates + (size_t)b * 512;
  float i = sigf(gr[n]);
  float f = sigf(gr[128 + n]);
  float gg = tanhf(gr[256 + n]);
  float o = sigf(gr[384 + n]);
  float c2 = f * c[g] + i * gg;
  c[g] = c2;
  h[g] = o * tanhf(c2);
}

// qq projection GEMM: out[b][o] = A[b][:] @ W[o][:] + bias[o]  (M=1024,N=128,K=128)
__global__ void qq_gemm_kernel(const float* A, const __bf16* W, const float* bias,
                               float* out) {
  __shared__ __bf16 At[16 * 32];
  int b0 = blockIdx.x * 16;
  int t = threadIdx.x, lane = t & 31, wave = t >> 5;
  int n0 = wave * 16;
  int sm = t >> 4, scol = (t & 15) * 2;
  v8f acc = {};
  for (int k0 = 0; k0 < HH; k0 += 32) {
    __syncthreads();
    stage_pair(At + sm * 32 + scol, &A[(size_t)(b0 + sm) * HH + k0 + scol]);
    __syncthreads();
    v16bf a = load_frag_A(At, lane);
    v16bf b = load_frag_B(W, HH, n0, k0, lane);
    acc = wmma_bf16(a, b, acc);
  }
  int n = n0 + (lane & 15), kh = lane >> 4;
  float bs = bias[n];
#pragma unroll
  for (int r = 0; r < 8; ++r)
    out[(size_t)(b0 + r + 8 * kh) * HH + n] = acc[r] + bs;
}

// ---------------------------------------------------------------------------
// Glimpse: u = mask? NEG : v.tanh(qq+e_g) ; softmax ; g_l = sum_l p[l]*e_g[l][:]
// one block per b (256 thr, 8 waves).
// ---------------------------------------------------------------------------
__global__ void glimpse_kernel(const float* qqg, const __bf16* eg, const float* gv,
                               const unsigned char* mask, float* g_l) {
  __shared__ float sq[128], sv[128], su[128], sred[8], part[256];
  int b = blockIdx.x, t = threadIdx.x, lane = t & 31, wave = t >> 5;
  if (t < 128) { sq[t] = qqg[b * HH + t]; sv[t] = gv[t]; }
  __syncthreads();
  const __bf16* eb = eg + (size_t)b * LL * HH;
  for (int li = 0; li < 16; ++li) {
    int l = wave * 16 + li;
    const __bf16* row = eb + l * HH;
    if (li < 15) __builtin_prefetch(row + HH, 0, 0);   // global_prefetch_b8
    float p = 0.f;
#pragma unroll
    for (int j = 0; j < 4; ++j) {
      int hh = lane * 4 + j;
      p += sv[hh] * tanhf(sq[hh] + (float)row[hh]);
    }
#pragma unroll
    for (int off = 16; off; off >>= 1) p += __shfl_xor(p, off, 32);
    if (lane == 0) su[l] = mask[b * LL + l] ? -1e9f : p;
  }
  __syncthreads();
  // softmax over su[0..127]
  float v = (t < 128) ? su[t] : -3.0e38f;
#pragma unroll
  for (int off = 16; off; off >>= 1) v = fmaxf(v, __shfl_xor(v, off, 32));
  if (lane == 0) sred[wave] = v;
  __syncthreads();
  float mx = sred[0];
#pragma unroll
  for (int w = 1; w < 8; ++w) mx = fmaxf(mx, sred[w]);
  float ex = (t < 128) ? __expf(su[t] - mx) : 0.f;
  float sm = ex;
#pragma unroll
  for (int off = 16; off; off >>= 1) sm += __shfl_xor(sm, off, 32);
  __syncthreads();
  if (lane == 0) sred[wave] = sm;
  __syncthreads();
  float tot = 0.f;
#pragma unroll
  for (int w = 0; w < 8; ++w) tot += sred[w];
  if (t < 128) su[t] = ex / tot;          // probabilities
  __syncthreads();
  // g_l[b][h] = sum_l p[l] * e_g[b][l][h]; two partial sums over l per h
  int hh = t & 127, half = t >> 7;
  float acc = 0.f;
  for (int l = half; l < LL; l += 2) acc += su[l] * (float)eb[l * HH + hh];
  part[t] = acc;
  __syncthreads();
  if (t < 128) g_l[b * HH + t] = part[t] + part[t + 128];
}

// ---------------------------------------------------------------------------
// Pointer head: logits = 10*tanh(v.tanh(qq_p+e_p)); mask; log_softmax ->
// d_out; record selection; update mask. one block per b.
// ---------------------------------------------------------------------------
__global__ void pointer_kernel(const float* qqp, const __bf16* ep, const float* pv,
                               const int* tours, unsigned char* mask,
                               float* out, int* outSel, int s) {
  __shared__ float sq[128], sv[128], su[128], sred[8];
  int b = blockIdx.x, t = threadIdx.x, lane = t & 31, wave = t >> 5;
  if (t < 128) { sq[t] = qqp[b * HH + t]; sv[t] = pv[t]; }
  __syncthreads();
  const __bf16* eb = ep + (size_t)b * LL * HH;
  for (int li = 0; li < 16; ++li) {
    int l = wave * 16 + li;
    const __bf16* row = eb + l * HH;
    if (li < 15) __builtin_prefetch(row + HH, 0, 0);
    float p = 0.f;
#pragma unroll
    for (int j = 0; j < 4; ++j) {
      int hh = lane * 4 + j;
      p += sv[hh] * tanhf(sq[hh] + (float)row[hh]);
    }
#pragma unroll
    for (int off = 16; off; off >>= 1) p += __shfl_xor(p, off, 32);
    if (lane == 0) su[l] = mask[b * LL + l] ? -1e9f : 10.0f * tanhf(p);
  }
  __syncthreads();
  float v = (t < 128) ? su[t] : -3.0e38f;
#pragma unroll
  for (int off = 16; off; off >>= 1) v = fmaxf(v, __shfl_xor(v, off, 32));
  if (lane == 0) sred[wave] = v;
  __syncthreads();
  float mx = sred[0];
#pragma unroll
  for (int w = 1; w < 8; ++w) mx = fmaxf(mx, sred[w]);
  float ex = (t < 128) ? __expf(su[t] - mx) : 0.f;
  float sm = ex;
#pragma unroll
  for (int off = 16; off; off >>= 1) sm += __shfl_xor(sm, off, 32);
  __syncthreads();
  if (lane == 0) sred[wave] = sm;
  __syncthreads();
  float tot = 0.f;
#pragma unroll
  for (int w = 0; w < 8; ++w) tot += sred[w];
  float lse = __logf(tot);
  if (t < 128)
    out[((size_t)b * LL + s) * LL + t] = su[t] - mx - lse;   // [B, step, L]
  if (t == 0) {
    int idx = tours[b * LL + s];
    outSel[b * LL + s] = idx;
    mask[b * LL + idx] = 1;
  }
}

__global__ void final_copy_kernel(const float* h, const float* c, float* out) {
  int g = blockIdx.x * blockDim.x + threadIdx.x;
  if (g < BB * HH) out[g] = h[g];
  else if (g < 2 * BB * HH) out[g] = c[g - BB * HH];
}

// ---------------------------------------------------------------------------
extern "C" void kernel_launch(void* const* d_in, const int* in_sizes, int n_in,
                              void* d_out, int out_size, void* d_ws, size_t ws_size,
                              hipStream_t stream) {
  const float* dec_in = (const float*)d_in[0];
  const float* emb    = (const float*)d_in[1];
  const float* h0     = (const float*)d_in[2];
  const float* c0     = (const float*)d_in[3];
  const float* ctx    = (const float*)d_in[4];
  const float* W_ih   = (const float*)d_in[5];
  const float* W_hh   = (const float*)d_in[6];
  const float* b_ih   = (const float*)d_in[7];
  const float* b_hh   = (const float*)d_in[8];
  const float* gWq    = (const float*)d_in[9];
  const float* gbq    = (const float*)d_in[10];
  const float* gWref  = (const float*)d_in[11];
  const float* gbref  = (const float*)d_in[12];
  const float* gv     = (const float*)d_in[13];
  const float* pWq    = (const float*)d_in[14];
  const float* pbq    = (const float*)d_in[15];
  const float* pWref  = (const float*)d_in[16];
  const float* pbref  = (const float*)d_in[17];
  const float* pv     = (const float*)d_in[18];
  const int*   tours  = (const int*)d_in[19];

  // workspace carve-up (~73 MB)
  char* w = (char*)d_ws;
  auto take = [&](size_t bytes) { char* p = w; w += (bytes + 255) & ~(size_t)255; return p; };
  __bf16* e_g   = (__bf16*)take((size_t)BB * LL * HH * 2);
  __bf16* e_p   = (__bf16*)take((size_t)BB * LL * HH * 2);
  __bf16* Wcat  = (__bf16*)take(512 * 256 * 2);
  __bf16* gWqb  = (__bf16*)take(HH * HH * 2);
  __bf16* pWqb  = (__bf16*)take(HH * HH * 2);
  __bf16* gWrefb= (__bf16*)take(HH * HH * 2);
  __bf16* pWrefb= (__bf16*)take(HH * HH * 2);
  float* h      = (float*)take((size_t)BB * HH * 4);
  float* c      = (float*)take((size_t)BB * HH * 4);
  float* gates  = (float*)take((size_t)BB * 512 * 4);
  float* qqg    = (float*)take((size_t)BB * HH * 4);
  float* qqp    = (float*)take((size_t)BB * HH * 4);
  float* g_l    = (float*)take((size_t)BB * HH * 4);
  unsigned char* mask = (unsigned char*)take((size_t)BB * LL);

  float* out_logp = (float*)d_out;                    // [B, L, L]
  int*   out_sel  = (int*)(out_logp + (size_t)BB * LL * LL);
  float* out_hc   = (float*)out_sel + (size_t)BB * LL;

  setup_kernel<<<2304, 256, 0, stream>>>(W_ih, W_hh, gWq, pWq, gWref, pWref, h0, c0,
                                         Wcat, gWqb, pWqb, gWrefb, pWrefb, h, c, mask);
  ref_proj_kernel<<<dim3(BB / 16, LL, 2), 256, 0, stream>>>(ctx, gWrefb, gbref,
                                                            pWrefb, pbref, e_g, e_p);
  for (int s = 0; s < LL; ++s) {
    lstm_gates_kernel<<<dim3(BB / 16, 4), 256, 0, stream>>>(dec_in, emb, tours, h,
                                                            Wcat, b_ih, b_hh, gates, s);
    lstm_pointwise_kernel<<<(BB * HH) / 256, 256, 0, stream>>>(gates, h, c);
    qq_gemm_kernel<<<BB / 16, 256, 0, stream>>>(h, gWqb, gbq, qqg);
    glimpse_kernel<<<BB, 256, 0, stream>>>(qqg, e_g, gv, mask, g_l);
    qq_gemm_kernel<<<BB / 16, 256, 0, stream>>>(g_l, pWqb, pbq, qqp);
    pointer_kernel<<<BB, 256, 0, stream>>>(qqp, e_p, pv, tours, mask,
                                           out_logp, out_sel, s);
  }
  final_copy_kernel<<<(2 * BB * HH) / 256, 256, 0, stream>>>(h, c, out_hc);
}
